// _GraphCoRelAdapter_29515015258494
// MI455X (gfx1250) — matrix operand order repaired and checked by reading, not compiled, not run on hardware
//
#include <hip/hip_runtime.h>
#include <hip/hip_bf16.h>

#define NPTS 16384
#define FDIM 64
#define HDIM 128
#define KNN  32
#define TEMP_INV 2.0f   // 1/TEMP, TEMP = 0.5
#define BIGF 1.0e30f

typedef __attribute__((ext_vector_type(16))) _Float16 v16h;
typedef __attribute__((ext_vector_type(8)))  float    v8f;

// ---------------------------------------------------------------------------
// WMMA fragment loaders (wave32), A and B in the same contiguous k order
// A (16x32 f16): lane m=l&15, g=l>>4, element j -> K = g*16 + j (same as B).
//   B (32x16 f16): lane n=l&15, g=l>>4,
// element j -> K = g*16 + j.   C/D f32: lane n, vgpr r -> M = r + 8*(l>>4).
// ---------------------------------------------------------------------------
__device__ __forceinline__ v16h load_A_frag(const _Float16* A, int lda, int kk) {
  const int lane = threadIdx.x & 31;
  const int m = lane & 15, g = lane >> 4;
  v16h a;
#pragma unroll
  for (int j = 0; j < 16; ++j) {
    const int k = kk + (g << 4) + j;
    a[j] = A[m * lda + k];
  }
  return a;
}

__device__ __forceinline__ v16h load_B_frag(const _Float16* B, int ldb, int kk) {
  const int lane = threadIdx.x & 31;
  const int n = lane & 15, g = lane >> 4;
  v16h b;
#pragma unroll
  for (int j = 0; j < 16; ++j) {
    const int k = kk + (g << 4) + j;
    b[j] = B[k * ldb + n];
  }
  return b;
}

// B = x^T tile: B[k][n] = xcols[n*FDIM + k]; contiguous in k per lane.
__device__ __forceinline__ v16h load_Bx_frag(const _Float16* xcols, int kk) {
  const int lane = threadIdx.x & 31;
  const int n = lane & 15, g = lane >> 4;
  const _Float16* p = xcols + n * FDIM + kk + (g << 4);
  v16h b;
#pragma unroll
  for (int j = 0; j < 16; ++j) b[j] = p[j];
  return b;
}

// ---------------------------------------------------------------------------
__global__ void f32_to_f16_kernel(const float* __restrict__ src,
                                  _Float16* __restrict__ dst, int n) {
  int i = blockIdx.x * blockDim.x + threadIdx.x;
  const int stride = gridDim.x * blockDim.x;
  for (; i < n; i += stride) dst[i] = (_Float16)src[i];
}

__global__ void row_sq_kernel(const float* __restrict__ x, float* __restrict__ sq) {
  __shared__ float r[64];
  const int row = blockIdx.x, t = threadIdx.x;
  const float v = x[(size_t)row * FDIM + t];
  r[t] = v * v;
  __syncthreads();
  for (int s = 32; s > 0; s >>= 1) {
    if (t < s) r[t] += r[t + s];
    __syncthreads();
  }
  if (t == 0) sq[row] = r[0];
}

// ---------------------------------------------------------------------------
// Fused distance-GEMM (WMMA) + top-K + softmax weights. 16 rows per block,
// 8 waves x 16 cols per tile iteration. All selection is done on SQUARED
// distances (sqrt is monotone) so the hot loop has no v_sqrt expansion;
// only 32 sqrts per row at softmax time. Threshold-filtered candidate queue
// in LDS (ds_add_rtn), merged by 16 row-owner threads.
// ---------------------------------------------------------------------------
__global__ void __launch_bounds__(256) knn_kernel(const _Float16* __restrict__ xh,
                                                  const float* __restrict__ sq,
                                                  int* __restrict__ out_idx,
                                                  float* __restrict__ out_w) {
  __shared__ _Float16 Atile[16 * FDIM];
  __shared__ float topd[16][KNN];   // squared distances
  __shared__ int   topi[16][KNN];
  __shared__ float thr[16];         // squared threshold (current kth max)
  __shared__ int   cnt[16];
  __shared__ float cd[16][128];
  __shared__ int   ci[16][128];

  const int row0 = blockIdx.x * 16;
  const int t = threadIdx.x;
  for (int i = t; i < 16 * FDIM; i += 256) Atile[i] = xh[(size_t)row0 * FDIM + i];
  if (t < 16) {
    thr[t] = BIGF;
    cnt[t] = 0;
    for (int k = 0; k < KNN; ++k) { topd[t][k] = BIGF; topi[t][k] = 0; }
  }
  __syncthreads();

  const int wave = t >> 5, lane = t & 31;
  const int n = lane & 15, g = lane >> 4;
  float sqr[8];
#pragma unroll
  for (int r = 0; r < 8; ++r) sqr[r] = sq[row0 + r + (g << 3)];

  for (int c0 = 0; c0 < NPTS; c0 += 128) {
    const int col0 = c0 + wave * 16;
    if (c0 + 128 < NPTS)  // speculative prefetch of next B tile -> global_prefetch_b8
      __builtin_prefetch(xh + (size_t)(col0 + 128) * FDIM + lane * 4, 0, 1);

    v8f c = {};
#pragma unroll
    for (int kk = 0; kk < FDIM; kk += 32) {
      v16h a = load_A_frag(Atile, FDIM, kk);
      v16h b = load_Bx_frag(xh + (size_t)col0 * FDIM, kk);
      c = __builtin_amdgcn_wmma_f32_16x16x32_f16(false, a, false, b, (short)0, c,
                                                 false, false);
    }
    const int col = col0 + n;
    const float sqc = sq[col];
#pragma unroll
    for (int r = 0; r < 8; ++r) {
      const int m = r + (g << 3);
      float d2 = fmaxf(sqr[r] + sqc - 2.0f * c[r], 0.0f);
      if (col == row0 + m) d2 = BIGF;  // exclude self
      if (d2 < thr[m]) {
        const int p = atomicAdd(&cnt[m], 1);   // LDS atomic (ds_add_rtn)
        if (p < 128) { cd[m][p] = d2; ci[m][p] = col; }
      }
    }
    __syncthreads();
    if (t < 16) {
      const int nc = cnt[t] < 128 ? cnt[t] : 128;
      float mv = thr[t];
      for (int q = 0; q < nc; ++q) {
        const float d2 = cd[t][q];
        if (d2 < mv) {
          int mp = 0;
          float cur = topd[t][0];
          for (int k = 1; k < KNN; ++k)
            if (topd[t][k] > cur) { cur = topd[t][k]; mp = k; }
          topd[t][mp] = d2;
          topi[t][mp] = ci[t][q];
          cur = topd[t][0];
          for (int k = 1; k < KNN; ++k) cur = fmaxf(cur, topd[t][k]);
          mv = cur;
        }
      }
      thr[t] = mv;
      cnt[t] = 0;
    }
    __syncthreads();
  }

  // softmax(-d/T) over the K kept neighbors (order irrelevant for agg);
  // only 32 sqrts per row happen here.
  if (t < 16) {
    float dmin2 = BIGF;
    for (int k = 0; k < KNN; ++k) dmin2 = fminf(dmin2, topd[t][k]);
    const float dmin = sqrtf(dmin2);
    float s = 0.0f;
    for (int k = 0; k < KNN; ++k)
      s += __expf((dmin - sqrtf(topd[t][k])) * TEMP_INV);
    const float inv = 1.0f / s;
    const int row = row0 + t;
    for (int k = 0; k < KNN; ++k) {
      out_w[(size_t)row * KNN + k] =
          __expf((dmin - sqrtf(topd[t][k])) * TEMP_INV) * inv;
      out_idx[(size_t)row * KNN + k] = topi[t][k];
    }
  }
}

// ---------------------------------------------------------------------------
// Generic WMMA GEMM: C[N,128] = act(A[N,Kd] @ B[Kd,128] + bias)
// Block: 16 rows, 8 waves (one 16x16 tile each). Kd multiple of 32.
// ---------------------------------------------------------------------------
template <bool SILU, bool OUT16>
__global__ void __launch_bounds__(256) gemm_wmma_kernel(
    const _Float16* __restrict__ A, const _Float16* __restrict__ B,
    const float* __restrict__ bias, int Kd, float* __restrict__ Cf,
    _Float16* __restrict__ Ch) {
  const int row0 = blockIdx.x * 16;
  const int wave = threadIdx.x >> 5;
  const int col0 = wave * 16;
  const _Float16* Ablk = A + (size_t)row0 * Kd;
  v8f c = {};
  for (int kk = 0; kk < Kd; kk += 32) {
    v16h a = load_A_frag(Ablk, Kd, kk);
    v16h b = load_B_frag(B + col0, HDIM, kk);
    c = __builtin_amdgcn_wmma_f32_16x16x32_f16(false, a, false, b, (short)0, c,
                                               false, false);
  }
  const int lane = threadIdx.x & 31;
  const int n = lane & 15, g = lane >> 4;
  const float bv = bias[col0 + n];
#pragma unroll
  for (int r = 0; r < 8; ++r) {
    const int m = r + (g << 3);
    float v = c[r] + bv;
    if (SILU) v = v / (1.0f + __expf(-v));
    if (OUT16)
      Ch[(size_t)(row0 + m) * HDIM + col0 + n] = (_Float16)v;
    else
      Cf[(size_t)(row0 + m) * HDIM + col0 + n] = v;
  }
}

// ---------------------------------------------------------------------------
// WMMA GEMM [N,128]@[128,128] fused with residual add + LayerNorm.
// Block holds full 16x128 rows -> LN reductions stay in LDS.
// ---------------------------------------------------------------------------
__global__ void __launch_bounds__(256) gemm_ln_kernel(
    const _Float16* __restrict__ A, const _Float16* __restrict__ B,
    const float* __restrict__ bias, const float* __restrict__ hin,
    const float* __restrict__ lng, const float* __restrict__ lnb,
    float* __restrict__ hout) {
  __shared__ float tile[16][HDIM];
  __shared__ float red[16][16];
  __shared__ float mu[16], rsd[16];
  const int row0 = blockIdx.x * 16;
  const int wave = threadIdx.x >> 5;
  const int col0 = wave * 16;
  const _Float16* Ablk = A + (size_t)row0 * HDIM;
  v8f c = {};
#pragma unroll
  for (int kk = 0; kk < HDIM; kk += 32) {
    v16h a = load_A_frag(Ablk, HDIM, kk);
    v16h b = load_B_frag(B + col0, HDIM, kk);
    c = __builtin_amdgcn_wmma_f32_16x16x32_f16(false, a, false, b, (short)0, c,
                                               false, false);
  }
  const int lane = threadIdx.x & 31;
  const int n = lane & 15, g = lane >> 4;
  const float bv = bias[col0 + n];
#pragma unroll
  for (int r = 0; r < 8; ++r) {
    const int m = r + (g << 3);
    tile[m][col0 + n] = c[r] + bv + hin[(size_t)(row0 + m) * HDIM + col0 + n];
  }
  __syncthreads();
  const int tr = threadIdx.x >> 4;  // row 0..15
  const int tc = threadIdx.x & 15;  // 8-col group 0..15
  float s = 0.0f;
#pragma unroll
  for (int j = 0; j < 8; ++j) s += tile[tr][tc * 8 + j];
  red[tr][tc] = s;
  __syncthreads();
  if (tc == 0) {
    float tot = 0.0f;
    for (int k = 0; k < 16; ++k) tot += red[tr][k];
    mu[tr] = tot * (1.0f / HDIM);
  }
  __syncthreads();
  const float m_ = mu[tr];
  float s2 = 0.0f;
#pragma unroll
  for (int j = 0; j < 8; ++j) {
    const float dd = tile[tr][tc * 8 + j] - m_;
    s2 += dd * dd;
  }
  red[tr][tc] = s2;
  __syncthreads();
  if (tc == 0) {
    float tot = 0.0f;
    for (int k = 0; k < 16; ++k) tot += red[tr][k];
    rsd[tr] = rsqrtf(tot * (1.0f / HDIM) + 1e-5f);
  }
  __syncthreads();
  const float rv = rsd[tr];
#pragma unroll
  for (int j = 0; j < 8; ++j) {
    const int h = tc * 8 + j;
    hout[(size_t)(row0 + tr) * HDIM + h] = (tile[tr][h] - m_) * rv * lng[h] + lnb[h];
  }
}

// ---------------------------------------------------------------------------
__global__ void build_updin_kernel(const float* __restrict__ hidden,
                                   const float* __restrict__ w,
                                   const int* __restrict__ idx,
                                   const _Float16* __restrict__ xh,
                                   _Float16* __restrict__ updin) {
  const int row = blockIdx.x, h = threadIdx.x;  // 128 threads
  __shared__ float sw[KNN];
  __shared__ int si[KNN];
  if (h < KNN) {
    sw[h] = w[(size_t)row * KNN + h];
    si[h] = idx[(size_t)row * KNN + h];
  }
  __syncthreads();
  float acc = 0.0f;
#pragma unroll 8
  for (int k = 0; k < KNN; ++k) acc += sw[k] * hidden[(size_t)si[k] * HDIM + h];
  updin[(size_t)row * 320 + h] = (_Float16)hidden[(size_t)row * HDIM + h];
  updin[(size_t)row * 320 + HDIM + h] = (_Float16)acc;
  if (h < FDIM) updin[(size_t)row * 320 + 2 * HDIM + h] = xh[(size_t)row * FDIM + h];
}

__global__ void build_roin_kernel(const float* __restrict__ hidden,
                                  const _Float16* __restrict__ xh,
                                  _Float16* __restrict__ roin) {
  const int row = blockIdx.x, h = threadIdx.x;  // 128 threads
  roin[(size_t)row * 192 + h] = (_Float16)hidden[(size_t)row * HDIM + h];
  if (h < FDIM) roin[(size_t)row * 192 + HDIM + h] = xh[(size_t)row * FDIM + h];
}

__global__ void readout_kernel(const _Float16* __restrict__ r1,
                               const float* __restrict__ w2,
                               const float* __restrict__ b2,
                               float* __restrict__ out) {
  __shared__ float red[128];
  const int row = blockIdx.x, h = threadIdx.x;
  red[h] = (float)r1[(size_t)row * HDIM + h] * w2[h];
  __syncthreads();
  for (int s = 64; s > 0; s >>= 1) {
    if (h < s) red[h] += red[h + s];
    __syncthreads();
  }
  if (h == 0) {
    const float v = red[0] + b2[0];
    out[row] = fmaxf(v, 0.0f) + log1pf(__expf(-fabsf(v)));  // stable softplus
  }
}

// ---------------------------------------------------------------------------
extern "C" void kernel_launch(void* const* d_in, const int* in_sizes, int n_in,
                              void* d_out, int out_size, void* d_ws, size_t ws_size,
                              hipStream_t stream) {
  (void)in_sizes; (void)n_in; (void)out_size; (void)ws_size;
  const float* x    = (const float*)d_in[0];
  const float* W_in = (const float*)d_in[1];
  const float* b_in = (const float*)d_in[2];
  const float* W_m1 = (const float*)d_in[3];
  const float* b_m1 = (const float*)d_in[4];
  const float* W_m2 = (const float*)d_in[5];
  const float* b_m2 = (const float*)d_in[6];
  const float* ln_g = (const float*)d_in[7];
  const float* ln_b = (const float*)d_in[8];
  const float* W_r1 = (const float*)d_in[9];
  const float* b_r1 = (const float*)d_in[10];
  const float* W_r2 = (const float*)d_in[11];
  const float* b_r2 = (const float*)d_in[12];
  float* out = (float*)d_out;

  char* ws = (char*)d_ws;
  size_t off = 0;
  auto alloc = [&](size_t bytes) -> void* {
    void* p = ws + off;
    off += (bytes + 255) & ~(size_t)255;
    return p;
  };
  _Float16* xh     = (_Float16*)alloc((size_t)NPTS * FDIM * 2);
  float*    sq     = (float*)   alloc((size_t)NPTS * 4);
  _Float16* Winh   = (_Float16*)alloc((size_t)FDIM * HDIM * 2);
  _Float16* Wm1h   = (_Float16*)alloc((size_t)320 * HDIM * 2);
  _Float16* Wm2h   = (_Float16*)alloc((size_t)HDIM * HDIM * 2);
  _Float16* Wr1h   = (_Float16*)alloc((size_t)192 * HDIM * 2);
  int*      kidx   = (int*)     alloc((size_t)NPTS * KNN * 4);
  float*    kw     = (float*)   alloc((size_t)NPTS * KNN * 4);
  float*    hidden = (float*)   alloc((size_t)NPTS * HDIM * 4);
  _Float16* updin  = (_Float16*)alloc((size_t)NPTS * 320 * 2);
  _Float16* m1h    = (_Float16*)alloc((size_t)NPTS * HDIM * 2);
  _Float16* roin   = (_Float16*)alloc((size_t)NPTS * 192 * 2);
  _Float16* r1h    = (_Float16*)alloc((size_t)NPTS * HDIM * 2);

  f32_to_f16_kernel<<<512, 256, 0, stream>>>(x, xh, NPTS * FDIM);
  f32_to_f16_kernel<<<32, 256, 0, stream>>>(W_in, Winh, FDIM * HDIM);
  f32_to_f16_kernel<<<160, 256, 0, stream>>>(W_m1, Wm1h, 320 * HDIM);
  f32_to_f16_kernel<<<64, 256, 0, stream>>>(W_m2, Wm2h, HDIM * HDIM);
  f32_to_f16_kernel<<<96, 256, 0, stream>>>(W_r1, Wr1h, 192 * HDIM);
  row_sq_kernel<<<NPTS, 64, 0, stream>>>(x, sq);

  // KNN depends only on x -> compute once, reuse for both message passes.
  knn_kernel<<<NPTS / 16, 256, 0, stream>>>(xh, sq, kidx, kw);

  gemm_wmma_kernel<false, false><<<NPTS / 16, 256, 0, stream>>>(
      xh, Winh, b_in, FDIM, hidden, ((_Float16*)nullptr));

  for (int step = 0; step < 2; ++step) {
    build_updin_kernel<<<NPTS, 128, 0, stream>>>(hidden, kw, kidx, xh, updin);
    gemm_wmma_kernel<true, true><<<NPTS / 16, 256, 0, stream>>>(
        updin, Wm1h, b_m1, 320, (float*)nullptr, m1h);
    gemm_ln_kernel<<<NPTS / 16, 256, 0, stream>>>(m1h, Wm2h, b_m2, hidden, ln_g,
                                                  ln_b, hidden);
  }

  build_roin_kernel<<<NPTS, 128, 0, stream>>>(hidden, xh, roin);
  gemm_wmma_kernel<true, true><<<NPTS / 16, 256, 0, stream>>>(
      roin, Wr1h, b_r1, 192, (float*)nullptr, r1h);
  readout_kernel<<<NPTS, 128, 0, stream>>>(r1h, W_r2, b_r2, out);
}
